// TransformerBlockQuantum_65481071402459
// MI455X (gfx1250) — matrix operand order repaired and checked by reading, not compiled, not run on hardware
//
#include <hip/hip_runtime.h>
#include <math.h>

typedef __attribute__((ext_vector_type(16))) _Float16 v16h;
typedef __attribute__((ext_vector_type(8)))  float    v8f;

union Frag16 { v16h v; uint4 q[2]; };

#define E_DIM   512
#define FFN_DIM 2048
#define NQ      8
#define ROWS    16          // tokens per workgroup
#define LDA     520         // f16 stride for A tile (512 + 8 pad), rows stay 16B aligned
#define LDC     516         // f32 stride for accumulate/LN tile
#define HLD     264         // f16 stride for h chunk (256 + 8 pad)
#define KCHUNK  256

__device__ __forceinline__ void load_a_frag(Frag16& a, const _Float16* arow, int koff, int hi) {
    a.q[0] = *(const uint4*)(arow + koff + hi * 8);
    a.q[1] = *(const uint4*)(arow + koff + 16 + hi * 8);
}
__device__ __forceinline__ void load_b_frag(Frag16& b, const _Float16* bbase, int koff, int hi) {
    const _Float16* p = bbase + koff + hi * 8;
    b.q[0] = *(const uint4*)(p);
    b.q[1] = *(const uint4*)(p + 16);
}

// ---------------------------------------------------------------- converter
__global__ void cvt_f32_to_f16(const float* __restrict__ src,
                               unsigned short* __restrict__ dst, int n) {
    int i = blockIdx.x * blockDim.x + threadIdx.x;
    if (i < n) {
        _Float16 h = (_Float16)src[i];
        ((_Float16*)dst)[i] = h;
    }
}

// ------------------------------------------------- kernel 1: attn + LN1 -> x1
__global__ __launch_bounds__(256)
void attn_ln1_kernel(const float* __restrict__ x,
                     const float* __restrict__ theta,        // 512 (H*DK flat)
                     const unsigned short* __restrict__ Wh_, // f16 [512][512] (f,e)
                     const float* __restrict__ bcomb,
                     const float* __restrict__ g1, const float* __restrict__ bln1,
                     float* __restrict__ x1) {
    const _Float16* __restrict__ Wh = (const _Float16*)Wh_;
    __shared__ __align__(16) _Float16 qh[ROWS * LDA];
    __shared__ __align__(16) float    accs[ROWS * LDC];

    const int tid = threadIdx.x;
    const int r0  = blockIdx.x * ROWS;

    // phase 1: q = cos(x + theta) -> f16 A tile in LDS
    for (int i = tid; i < ROWS * E_DIM; i += 256) {
        int m = i >> 9, e = i & 511;
        float v = cosf(x[(size_t)(r0 + m) * E_DIM + e] + theta[e]);
        qh[m * LDA + e] = (_Float16)v;
    }
    __syncthreads();

    const int wave = tid >> 5;
    const int lane = tid & 31;
    const int lo16 = lane & 15;
    const int hi   = lane >> 4;

    v8f acc[4] = {};   // 4 col tiles of 16, cols [wave*64, wave*64+64)

    const _Float16* arow = &qh[lo16 * LDA];
    const _Float16* bbase[4];
#pragma unroll
    for (int j = 0; j < 4; ++j)
        bbase[j] = Wh + (size_t)(wave * 64 + j * 16 + lo16) * E_DIM;

    // software-pipelined K loop: double-buffered A (LDS) + B (global)
    Frag16 a[2], b[2][4];
    load_a_frag(a[0], arow, 0, hi);
#pragma unroll
    for (int j = 0; j < 4; ++j) load_b_frag(b[0][j], bbase[j], 0, hi);

#pragma unroll
    for (int kk = 0; kk < E_DIM; kk += 32) {
        const int cur = (kk >> 5) & 1, nxt = cur ^ 1;
        const int kn = kk + 32;
        if (kn < E_DIM) {
            load_a_frag(a[nxt], arow, kn, hi);
#pragma unroll
            for (int j = 0; j < 4; ++j) load_b_frag(b[nxt][j], bbase[j], kn, hi);
        }
#pragma unroll
        for (int j = 0; j < 4; ++j)
            acc[j] = __builtin_amdgcn_wmma_f32_16x16x32_f16(
                false, a[cur].v, false, b[cur][j].v, (short)0, acc[j], false, false);
    }

    // bias + residual into LDS accumulate tile
#pragma unroll
    for (int j = 0; j < 4; ++j) {
        int n = wave * 64 + j * 16 + lo16;
        float bc = bcomb[n];
#pragma unroll
        for (int v = 0; v < 8; ++v) {
            int m = v + hi * 8;
            accs[m * LDC + n] = acc[j][v] + bc + x[(size_t)(r0 + m) * E_DIM + n];
        }
    }
    __syncthreads();

    // layernorm per row (wave handles 2 rows), write x1
    for (int rr = 0; rr < 2; ++rr) {
        int m = wave * 2 + rr;
        float s = 0.f, ss = 0.f;
        for (int e = lane; e < E_DIM; e += 32) {
            float v = accs[m * LDC + e];
            s += v; ss += v * v;
        }
#pragma unroll
        for (int off = 16; off > 0; off >>= 1) {
            s  += __shfl_xor(s, off, 32);
            ss += __shfl_xor(ss, off, 32);
        }
        float mean = s * (1.f / (float)E_DIM);
        float var  = ss * (1.f / (float)E_DIM) - mean * mean;
        float inv  = rsqrtf(var + 1e-5f);
        for (int e = lane; e < E_DIM; e += 32) {
            float v = (accs[m * LDC + e] - mean) * inv * g1[e] + bln1[e];
            x1[(size_t)(r0 + m) * E_DIM + e] = v;
        }
    }
}

// --------------------------- kernel 2: quantum FFN (fused GEMM2+3) + LN2 -> out
__global__ __launch_bounds__(256)
void ffn_ln2_kernel(const float* __restrict__ x1,
                    const float* __restrict__ thetaf,        // 8
                    const float* __restrict__ W1,            // [2048][8]
                    const float* __restrict__ b1v,           // 2048
                    const unsigned short* __restrict__ W2h_, // f16 [512][2048] (e,f)
                    const float* __restrict__ b2v,           // 512
                    const float* __restrict__ g2, const float* __restrict__ bln2,
                    float* __restrict__ out) {
    const _Float16* __restrict__ W2h = (const _Float16*)W2h_;
    __shared__ __align__(16) float    fq[ROWS][NQ];
    __shared__ __align__(16) _Float16 hh[ROWS * HLD];
    __shared__ __align__(16) float    accs[ROWS * LDC];

    const int tid = threadIdx.x;
    const int r0  = blockIdx.x * ROWS;
    const int wave = tid >> 5, lane = tid & 31, lo16 = lane & 15, hi = lane >> 4;

    if (tid < ROWS * NQ) {
        int m = tid >> 3, qi = tid & 7;
        fq[m][qi] = cosf(x1[(size_t)(r0 + m) * E_DIM + qi] + thetaf[qi]);
    }

    v8f acc[4] = {};
    const _Float16* arow = &hh[lo16 * HLD];
    const _Float16* bbase0[4];
#pragma unroll
    for (int j = 0; j < 4; ++j)
        bbase0[j] = W2h + (size_t)(wave * 64 + j * 16 + lo16) * FFN_DIM;

    for (int c = 0; c < FFN_DIM / KCHUNK; ++c) {
        __syncthreads();   // protect hh (and fq on first iter)
        // build h chunk: thread owns column f = c*256 + tid
        {
            int f = c * KCHUNK + tid;
            const float* wr = W1 + (size_t)f * NQ;
            float w0 = wr[0], w1 = wr[1], w2 = wr[2], w3 = wr[3];
            float w4 = wr[4], w5 = wr[5], w6 = wr[6], w7 = wr[7];
            float bb = b1v[f];
#pragma unroll
            for (int m = 0; m < ROWS; ++m) {
                const float* fm = fq[m];
                float s = bb + fm[0] * w0 + fm[1] * w1 + fm[2] * w2 + fm[3] * w3
                             + fm[4] * w4 + fm[5] * w5 + fm[6] * w6 + fm[7] * w7;
                hh[m * HLD + tid] = (_Float16)fmaxf(s, 0.f);
            }
        }
        __syncthreads();

        // WMMA over this 256-wide K chunk, double-buffered A (LDS) + B (global)
        const int kb = c * KCHUNK;
        Frag16 a[2], b[2][4];
        load_a_frag(a[0], arow, 0, hi);
#pragma unroll
        for (int j = 0; j < 4; ++j) load_b_frag(b[0][j], bbase0[j], kb, hi);

#pragma unroll
        for (int kk = 0; kk < KCHUNK; kk += 32) {
            const int cur = (kk >> 5) & 1, nxt = cur ^ 1;
            const int kn = kk + 32;
            if (kn < KCHUNK) {
                load_a_frag(a[nxt], arow, kn, hi);
#pragma unroll
                for (int j = 0; j < 4; ++j) load_b_frag(b[nxt][j], bbase0[j], kb + kn, hi);
            }
#pragma unroll
            for (int j = 0; j < 4; ++j)
                acc[j] = __builtin_amdgcn_wmma_f32_16x16x32_f16(
                    false, a[cur].v, false, b[cur][j].v, (short)0, acc[j], false, false);
        }
    }

    // bias + residual
#pragma unroll
    for (int j = 0; j < 4; ++j) {
        int n = wave * 64 + j * 16 + lo16;
        float bc = b2v[n];
#pragma unroll
        for (int v = 0; v < 8; ++v) {
            int m = v + hi * 8;
            accs[m * LDC + n] = acc[j][v] + bc + x1[(size_t)(r0 + m) * E_DIM + n];
        }
    }
    __syncthreads();

    // LN2 -> final output
    for (int rr = 0; rr < 2; ++rr) {
        int m = wave * 2 + rr;
        float s = 0.f, ss = 0.f;
        for (int e = lane; e < E_DIM; e += 32) {
            float v = accs[m * LDC + e];
            s += v; ss += v * v;
        }
#pragma unroll
        for (int off = 16; off > 0; off >>= 1) {
            s  += __shfl_xor(s, off, 32);
            ss += __shfl_xor(ss, off, 32);
        }
        float mean = s * (1.f / (float)E_DIM);
        float var  = ss * (1.f / (float)E_DIM) - mean * mean;
        float inv  = rsqrtf(var + 1e-5f);
        for (int e = lane; e < E_DIM; e += 32) {
            float v = (accs[m * LDC + e] - mean) * inv * g2[e] + bln2[e];
            out[(size_t)(r0 + m) * E_DIM + e] = v;
        }
    }
}

// ---------------------------------------------------------------- launcher
extern "C" void kernel_launch(void* const* d_in, const int* in_sizes, int n_in,
                              void* d_out, int out_size, void* d_ws, size_t ws_size,
                              hipStream_t stream) {
    const float* x       = (const float*)d_in[0];   // [8,4096,512]
    const float* th_attn = (const float*)d_in[1];   // [64,8] == flat 512
    const float* W_comb  = (const float*)d_in[2];   // [512,512]
    const float* b_comb  = (const float*)d_in[3];   // [512]
    const float* ln1_g   = (const float*)d_in[4];
    const float* ln1_b   = (const float*)d_in[5];
    const float* th_ffn  = (const float*)d_in[6];   // [8]
    const float* W1      = (const float*)d_in[7];   // [2048,8]
    const float* b1      = (const float*)d_in[8];   // [2048]
    const float* W2      = (const float*)d_in[9];   // [512,2048]
    const float* b2      = (const float*)d_in[10];  // [512]
    const float* ln2_g   = (const float*)d_in[11];
    const float* ln2_b   = (const float*)d_in[12];
    float* out = (float*)d_out;

    const int Ntok = 8 * 4096;                      // 32768 tokens
    char* ws = (char*)d_ws;
    float*          x1  = (float*)ws;                                       // 64 MB
    unsigned short* Wch = (unsigned short*)(ws + (size_t)64 * 1024 * 1024); // 512 KB
    unsigned short* W2h = (unsigned short*)(ws + (size_t)65 * 1024 * 1024); // 2 MB

    // convert weights to f16 (B-matrix operands)
    cvt_f32_to_f16<<<(512 * 512 + 255) / 256, 256, 0, stream>>>(W_comb, Wch, 512 * 512);
    cvt_f32_to_f16<<<(512 * 2048 + 255) / 256, 256, 0, stream>>>(W2, W2h, 512 * 2048);

    const int nblk = Ntok / ROWS;   // 2048 workgroups
    attn_ln1_kernel<<<nblk, 256, 0, stream>>>(x, th_attn, Wch, b_comb, ln1_g, ln1_b, x1);
    ffn_ln2_kernel<<<nblk, 256, 0, stream>>>(x1, th_ffn, W1, b1, W2h, b2, ln2_g, ln2_b, out);
    (void)in_sizes; (void)n_in; (void)out_size; (void)ws_size;
}
